// MACE_30863634989146
// MI455X (gfx1250) — compile-verified, matrix operand order checked by reading, not run on hardware
//
#include <hip/hip_runtime.h>

typedef __attribute__((ext_vector_type(16))) _Float16 v16h;
typedef __attribute__((ext_vector_type(8)))  _Float16 v8h;
typedef __attribute__((ext_vector_type(8)))  float    v8f;

#define N_NODES 20000
#define N_EDGES 320000
#define PI_F 3.14159265358979323846f

// fast silu: x * v_rcp_f32(1 + exp(-x))  (avoids IEEE div macro-expansion)
__device__ __forceinline__ float silu_f(float x) {
    return x * __builtin_amdgcn_rcpf(1.0f + __expf(-x));
}
// k index inside a 16x32 f16 A/B fragment for element e, lane-half sel
__device__ __forceinline__ int frag_k(int e, int sel) {
    return (e & 7) | (sel << 3) | ((e & 8) << 1);
}

// ---------------------------------------------------------------- weights -> f16
__global__ __launch_bounds__(256)
void prep_weights_kernel(const float* __restrict__ rW1, const float* __restrict__ rW2,
                         const float* __restrict__ Wupd,
                         _Float16* __restrict__ W1h, _Float16* __restrict__ W2h,
                         _Float16* __restrict__ Wupdh)
{
    int tid = blockIdx.x * 256 + threadIdx.x;
    if (tid < 2 * 8 * 64)    W1h[tid]   = (_Float16)rW1[tid];
    if (tid < 2 * 64 * 128)  W2h[tid]   = (_Float16)rW2[tid];
    if (tid < 2 * 128 * 32)  Wupdh[tid] = (_Float16)Wupd[tid];
}

// ---------------------------------------------------------------- zero agg
__global__ __launch_bounds__(256)
void zero_kernel(float4* __restrict__ p, int n4)
{
    int i = blockIdx.x * 256 + threadIdx.x;
    if (i < n4) p[i] = make_float4(0.f, 0.f, 0.f, 0.f);
}

// ---------------------------------------------------------------- h init
__global__ __launch_bounds__(256)
void h_init_kernel(const float* __restrict__ embed, const int* __restrict__ specie,
                   float* __restrict__ h, int N)
{
    int p = blockIdx.x * 256 + threadIdx.x;
    if (p >= N * 32) return;
    h[p] = embed[specie[p >> 5] * 32 + (p & 31)];
}

// ---------------------------------------------------------------- edge geometry
__global__ __launch_bounds__(256)
void edge_geom_kernel(const float* __restrict__ vec,
                      _Float16* __restrict__ radial_h,
                      float* __restrict__ sh, int E)
{
    int e = blockIdx.x * 256 + threadIdx.x;
    if (e >= E) return;
    float x = vec[3 * e + 0], y = vec[3 * e + 1], z = vec[3 * e + 2];
    float r2 = x * x + y * y + z * z + 1e-12f;
    float inv_r = __builtin_amdgcn_rsqf(r2);     // v_rsq_f32: 1/sqrt(r2)
    float r = r2 * inv_r;                        // = sqrt(r2)
    float xr = r * 0.2f;                         // r / R_MAX
    float env = 0.0f;
    if (xr < 1.0f) {
        float x2 = xr * xr, x4 = x2 * x2, x6 = x4 * x2, x7 = x6 * xr, x8 = x4 * x4;
        env = 1.0f - 28.0f * x6 + 48.0f * x7 - 21.0f * x8;
    }
    float coef = 0.6324555320336759f * env * inv_r;  // sqrt(2/R_MAX) * env / r
    v8h rv;
#pragma unroll
    for (int n = 0; n < 8; ++n)
        rv[n] = (_Float16)(coef * __sinf((float)(n + 1) * PI_F * xr));
    *(v8h*)(radial_h + (long)e * 8) = rv;

    float ux = x * inv_r, uy = y * inv_r, uz = z * inv_r;
    const float s3 = 1.7320508075688772f, s5 = 2.2360679774997896f, s15 = 3.872983346207417f;
    const float ca = 2.0916500663351889f;   // sqrt(35/8)
    const float cb = 10.246950765959598f;   // sqrt(105)
    const float cc = 1.6201851746019651f;   // sqrt(21/8)
    const float cd = 2.6457513110645907f;   // sqrt(7)
    float o[16];
    o[0]  = 1.0f;
    o[1]  = s3 * ux; o[2] = s3 * uy; o[3] = s3 * uz;
    o[4]  = s15 * ux * uy; o[5] = s15 * uy * uz;
    o[6]  = 0.5f * s5 * (3.0f * uz * uz - 1.0f);
    o[7]  = s15 * ux * uz;
    o[8]  = 0.5f * s15 * (ux * ux - uy * uy);
    o[9]  = ca * uy * (3.0f * ux * ux - uy * uy);
    o[10] = cb * ux * uy * uz;
    o[11] = cc * uy * (5.0f * uz * uz - 1.0f);
    o[12] = 0.5f * cd * (5.0f * uz * uz * uz - 3.0f * uz);
    o[13] = cc * ux * (5.0f * uz * uz - 1.0f);
    o[14] = 0.5f * cb * uz * (ux * ux - uy * uy);
    o[15] = ca * ux * (ux * ux - 3.0f * uy * uy);
    float4* sp = (float4*)(sh + (long)e * 16);
    sp[0] = make_float4(o[0], o[1], o[2], o[3]);
    sp[1] = make_float4(o[4], o[5], o[6], o[7]);
    sp[2] = make_float4(o[8], o[9], o[10], o[11]);
    sp[3] = make_float4(o[12], o[13], o[14], o[15]);
}

// ---------------------------------------------------------------- fused edge layer
// 128 threads = 4 waves; each wave owns a 16-edge tile (WMMA M=16).
__global__ __launch_bounds__(128)
void edge_layer_kernel(const _Float16* __restrict__ radial_h,
                       const float* __restrict__ sh_g,
                       const float* __restrict__ h_in,
                       const _Float16* __restrict__ W1h,   // 8 x 64
                       const _Float16* __restrict__ W2h,   // 64 x 128
                       const int* __restrict__ senders,
                       const int* __restrict__ receivers,
                       float* __restrict__ agg)
{
    __shared__ _Float16 lW1f[4 * 32 * 16];          // pre-fragmented B for GEMM1 (4KB)
    __shared__ _Float16 lW2f[2 * 8 * 32 * 16];      // pre-fragmented B for GEMM2 (16KB)
    __shared__ _Float16 hidbuf[4][16 * 64];         // per-wave GEMM1->GEMM2 transpose (8KB)
    __shared__ float    wbuf[4][16 * 128];          // per-wave w tile (32KB)
    __shared__ float    shbuf[4][16 * 16];          // per-wave sh tile (4KB)

    const int tid = threadIdx.x;
    const int lane = tid & 31;
    const int wave = tid >> 5;
    const int n0 = lane & 15;
    const int sel = lane >> 4;

    // ---- stage pre-fragmented weights (cooperative, then barrier)
    for (int idx = tid; idx < 4 * 32 * 16; idx += 128) {
        int e = idx & 15, l = (idx >> 4) & 31, t = idx >> 9;
        int k = frag_k(e, l >> 4);
        int n = t * 16 + (l & 15);
        lW1f[idx] = (k < 8) ? W1h[k * 64 + n] : (_Float16)0.0f;  // K padded 8->32
    }
    for (int idx = tid; idx < 2 * 8 * 32 * 16; idx += 128) {
        int e = idx & 15, l = (idx >> 4) & 31, j = (idx >> 9) & 7, kc = idx >> 12;
        int k = kc * 32 + frag_k(e, l >> 4);
        int n = j * 16 + (l & 15);
        lW2f[idx] = W2h[k * 128 + n];
    }
    __syncthreads();

    const long ebase = ((long)blockIdx.x * 4 + wave) * 16;

    // ---- stage sh tile (16 edges x 16) into LDS; prefetch sender h rows
    {
        const float4* src = (const float4*)(sh_g + ebase * 16);
        float4* dst = (float4*)shbuf[wave];
        dst[lane] = src[lane];
        dst[lane + 32] = src[lane + 32];
        if (lane < 16) {
            int s = senders[ebase + lane];
            __builtin_prefetch(h_in + (long)s * 32, 0, 1);   // global_prefetch
        }
    }

    // ---- A fragment for GEMM1: 16 edges x 8 radial (K padded to 32 with zeros)
    v16h a1 = {};
    if (lane < 16) {
        v8h rv = *(const v8h*)(radial_h + (ebase + lane) * 8);
#pragma unroll
        for (int e = 0; e < 8; ++e) a1[e] = rv[e];   // K=0..7 live, rest zero
    }

    // ---- GEMM1: (16x32) x (32x64) -> 4 accumulator tiles
    v8f acc1[4];
#pragma unroll
    for (int t = 0; t < 4; ++t) {
        v16h b = *(const v16h*)(lW1f + (t * 32 + lane) * 16);
        v8f c = {};
        acc1[t] = __builtin_amdgcn_wmma_f32_16x16x32_f16(
            false, a1, false, b, (short)0, c, false, false);
    }

    // ---- silu + transpose through LDS into GEMM2 A layout (f16)
#pragma unroll
    for (int t = 0; t < 4; ++t) {
        int n = t * 16 + n0;
#pragma unroll
        for (int r = 0; r < 8; ++r)
            hidbuf[wave][(r + sel * 8) * 64 + n] = (_Float16)silu_f(acc1[t][r]);
    }
    asm volatile("" ::: "memory");   // keep ds_store -> ds_load order (HW is in-order per wave)

    v16h a2[2];
#pragma unroll
    for (int kc = 0; kc < 2; ++kc) {
        const _Float16* p = hidbuf[wave] + n0 * 64 + kc * 32 + sel * 8;
        v8h lo = *(const v8h*)p;
        v8h hi = *(const v8h*)(p + 16);
        v16h a;
#pragma unroll
        for (int e = 0; e < 8; ++e) { a[e] = lo[e]; a[e + 8] = hi[e]; }
        a2[kc] = a;
    }

    // ---- GEMM2: (16x64) x (64x128) -> w tile, dumped to LDS in edge-major form
#pragma unroll
    for (int j = 0; j < 8; ++j) {
        v8f c = {};
        v16h b0 = *(const v16h*)(lW2f + ((0 * 8 + j) * 32 + lane) * 16);
        c = __builtin_amdgcn_wmma_f32_16x16x32_f16(false, a2[0], false, b0, (short)0, c, false, false);
        v16h b1 = *(const v16h*)(lW2f + ((1 * 8 + j) * 32 + lane) * 16);
        c = __builtin_amdgcn_wmma_f32_16x16x32_f16(false, a2[1], false, b1, (short)0, c, false, false);
        int n = j * 16 + n0;
#pragma unroll
        for (int r = 0; r < 8; ++r)
            wbuf[wave][(r + sel * 8) * 128 + n] = c[r];
    }
    asm volatile("" ::: "memory");

    // ---- message + scatter: lane = channel; agg is 41MB -> L2-resident atomics
#pragma unroll 1
    for (int i = 0; i < 16; ++i) {
        long edge = ebase + i;
        int snd = senders[edge];
        int rcv = receivers[edge];
        float hs = h_in[(long)snd * 32 + lane];
        float4 wv = *(const float4*)&wbuf[wave][i * 128 + lane * 4];
        const float* shv = shbuf[wave] + i * 16;
        float* base = agg + (long)rcv * 512 + lane * 16;
        float m0 = wv.x * hs, m1 = wv.y * hs, m2 = wv.z * hs, m3 = wv.w * hs;
        unsafeAtomicAdd(base + 0, m0 * shv[0]);
#pragma unroll
        for (int k = 1; k < 4; ++k)  unsafeAtomicAdd(base + k, m1 * shv[k]);
#pragma unroll
        for (int k = 4; k < 9; ++k)  unsafeAtomicAdd(base + k, m2 * shv[k]);
#pragma unroll
        for (int k = 9; k < 16; ++k) unsafeAtomicAdd(base + k, m3 * shv[k]);
    }
}

// ---------------------------------------------------------------- node update (WMMA)
// 256 threads = 8 waves; block handles 128 nodes (wave -> 16-node WMMA tile)
__global__ __launch_bounds__(256)
void node_update_kernel(const float* __restrict__ agg,
                        const _Float16* __restrict__ Wupdh,   // 128 x 32
                        float* __restrict__ h_out, int N)
{
    __shared__ _Float16 scalbuf[128 * 128];     // 32KB
    __shared__ _Float16 lWf[8 * 32 * 16];       // 8KB: frags (kc,j)
    const int tid = threadIdx.x;
    const int lane = tid & 31;
    const int wave = tid >> 5;
    const int n0 = lane & 15;
    const int sel = lane >> 4;

    for (int idx = tid; idx < 8 * 32 * 16; idx += 256) {
        int e = idx & 15, l = (idx >> 4) & 31, j = (idx >> 9) & 1, kc = idx >> 10;
        int k = kc * 32 + frag_k(e, l >> 4);
        int n = j * 16 + (l & 15);
        lWf[idx] = Wupdh[k * 32 + n];
    }

    const int nb = blockIdx.x * 128;
    const float s = 1.0f / 16.0f;               // 1 / AVG_NEIGH
#pragma unroll 1
    for (int it = 0; it < 16; ++it) {
        int p = it * 256 + tid;
        int nl = p >> 5, c = p & 31;
        int node = min(nb + nl, N - 1);
        const float4* ap = (const float4*)(agg + (long)node * 512 + c * 16);
        float4 a0 = ap[0], a1 = ap[1], a2 = ap[2], a3 = ap[3];
        float v0 = a0.x * s;
        float q1 = a0.y * a0.y + a0.z * a0.z + a0.w * a0.w;
        float q2 = a1.x * a1.x + a1.y * a1.y + a1.z * a1.z + a1.w * a1.w + a2.x * a2.x;
        float q3 = a2.y * a2.y + a2.z * a2.z + a2.w * a2.w +
                   a3.x * a3.x + a3.y * a3.y + a3.z * a3.z + a3.w * a3.w;
        scalbuf[nl * 128 + c]       = (_Float16)v0;
        scalbuf[nl * 128 + 32 + c]  = (_Float16)__builtin_amdgcn_sqrtf(q1 * s * s + 1e-12f);
        scalbuf[nl * 128 + 64 + c]  = (_Float16)__builtin_amdgcn_sqrtf(q2 * s * s + 1e-12f);
        scalbuf[nl * 128 + 96 + c]  = (_Float16)__builtin_amdgcn_sqrtf(q3 * s * s + 1e-12f);
    }
    __syncthreads();

    // wave -> nodes [nb + 16*wave, +16): (16x128) x (128x32) WMMA
    v16h af[4];
#pragma unroll
    for (int kc = 0; kc < 4; ++kc) {
        const _Float16* p = scalbuf + (wave * 16 + n0) * 128 + kc * 32 + sel * 8;
        v8h lo = *(const v8h*)p;
        v8h hi = *(const v8h*)(p + 16);
        v16h a;
#pragma unroll
        for (int e = 0; e < 8; ++e) { a[e] = lo[e]; a[e + 8] = hi[e]; }
        af[kc] = a;
    }
#pragma unroll
    for (int j = 0; j < 2; ++j) {
        v8f c = {};
#pragma unroll
        for (int kc = 0; kc < 4; ++kc) {
            v16h b = *(const v16h*)(lWf + ((kc * 2 + j) * 32 + lane) * 16);
            c = __builtin_amdgcn_wmma_f32_16x16x32_f16(false, af[kc], false, b, (short)0, c, false, false);
        }
#pragma unroll
        for (int r = 0; r < 8; ++r) {
            int node = nb + wave * 16 + r + sel * 8;
            if (node < N)
                h_out[(long)node * 32 + j * 16 + n0] = silu_f(c[r]);
        }
    }
}

// ---------------------------------------------------------------- readout
__global__ __launch_bounds__(256)
void readout_kernel(const float* __restrict__ h, const float* __restrict__ Wro,
                    const float* __restrict__ Wout, float* __restrict__ out, int N)
{
    __shared__ float lro[32 * 16];
    __shared__ float lo[16];
    int tid = threadIdx.x;
    for (int i = tid; i < 512; i += 256) lro[i] = Wro[i];
    if (tid < 16) lo[tid] = Wout[tid];
    __syncthreads();
    int node = blockIdx.x * 256 + tid;
    if (node >= N) return;
    float hv[32];
    const float4* hp = (const float4*)(h + (long)node * 32);
#pragma unroll
    for (int q = 0; q < 8; ++q) {
        float4 v = hp[q];
        hv[q * 4 + 0] = v.x; hv[q * 4 + 1] = v.y; hv[q * 4 + 2] = v.z; hv[q * 4 + 3] = v.w;
    }
    float o = 0.0f;
#pragma unroll
    for (int j = 0; j < 16; ++j) {
        float t = 0.0f;
#pragma unroll
        for (int c = 0; c < 32; ++c) t += hv[c] * lro[c * 16 + j];
        o += silu_f(t) * lo[j];
    }
    out[node] = o;
}

// ---------------------------------------------------------------- launch
extern "C" void kernel_launch(void* const* d_in, const int* in_sizes, int n_in,
                              void* d_out, int out_size, void* d_ws, size_t ws_size,
                              hipStream_t stream)
{
    const float* vectors  = (const float*)d_in[0];
    const float* embed    = (const float*)d_in[1];
    const float* rW1      = (const float*)d_in[2];
    const float* rW2      = (const float*)d_in[3];
    const float* Wupd     = (const float*)d_in[4];
    const float* Wro      = (const float*)d_in[5];
    const float* Wout     = (const float*)d_in[6];
    const int*   specie   = (const int*)d_in[7];
    const int*   senders  = (const int*)d_in[8];
    const int*   recvs    = (const int*)d_in[9];
    float* out = (float*)d_out;

    char* ws = (char*)d_ws;
    size_t off = 0;
    auto take = [&](size_t bytes) -> void* {
        void* p = ws + off;
        off += (bytes + 255) & ~(size_t)255;
        return p;
    };
    _Float16* radial_h = (_Float16*)take((size_t)N_EDGES * 8 * 2);   //  5.1 MB
    float*    sh       = (float*)take((size_t)N_EDGES * 16 * 4);     // 20.5 MB
    float*    hA       = (float*)take((size_t)N_NODES * 32 * 4);     //  2.6 MB
    float*    hB       = (float*)take((size_t)N_NODES * 32 * 4);     //  2.6 MB
    float*    agg      = (float*)take((size_t)N_NODES * 512 * 4);    // 41.0 MB (L2-resident)
    _Float16* W1h      = (_Float16*)take(2 * 8 * 64 * 2);
    _Float16* W2h      = (_Float16*)take(2 * 64 * 128 * 2);
    _Float16* Wupdh    = (_Float16*)take(2 * 128 * 32 * 2);

    prep_weights_kernel<<<64, 256, 0, stream>>>(rW1, rW2, Wupd, W1h, W2h, Wupdh);
    edge_geom_kernel<<<N_EDGES / 256, 256, 0, stream>>>(vectors, radial_h, sh, N_EDGES);
    h_init_kernel<<<(N_NODES * 32 + 255) / 256, 256, 0, stream>>>(embed, specie, hA, N_NODES);

    const int agg4 = N_NODES * 512 / 4;
    for (int layer = 0; layer < 2; ++layer) {
        zero_kernel<<<(agg4 + 255) / 256, 256, 0, stream>>>((float4*)agg, agg4);
        const float* hin = (layer == 0) ? hA : hB;
        float* hout = (layer == 0) ? hB : hA;
        edge_layer_kernel<<<N_EDGES / 64, 128, 0, stream>>>(
            radial_h, sh, hin, W1h + layer * 512, W2h + layer * 8192,
            senders, recvs, agg);
        node_update_kernel<<<(N_NODES + 127) / 128, 256, 0, stream>>>(
            agg, Wupdh + layer * 4096, hout, N_NODES);
    }
    readout_kernel<<<(N_NODES + 255) / 256, 256, 0, stream>>>(hA, Wro, Wout, out, N_NODES);
}
